// CausalSelfAttention_111669150062
// MI455X (gfx1250) — compile-verified, matrix operand order checked by reading, not run on hardware
//
#include <hip/hip_runtime.h>
#include <math.h>

// ---------------------------------------------------------------------------
// CDNA5 (gfx1250) attention block: bf16 WMMA GEMMs + flash attention.
// B=4, T=2048, C=2048, H=16, D=128. wave32 only.
// Compute-bound (~412 GFLOP vs ~300MB HBM traffic) -> everything through
// v_wmma_f32_16x16x32_bf16; tile staging via GLOBAL_LOAD_ASYNC_TO_LDS when
// the toolchain exposes it (ASYNCcnt path), VGPR round-trip otherwise.
// ---------------------------------------------------------------------------

typedef __bf16 bf16;
typedef __bf16  v16bf __attribute__((ext_vector_type(16)));
typedef float   v8f   __attribute__((ext_vector_type(8)));

#if defined(__AMDGCN__) && defined(__has_builtin)
#if __has_builtin(__builtin_amdgcn_global_load_async_to_lds_b128)
#define USE_ASYNC_LDS 1
#endif
#endif
#ifndef USE_ASYNC_LDS
#define USE_ASYNC_LDS 0
#endif

#if USE_ASYNC_LDS
// Async memory->LDS copy, 16B per lane, no VGPR data round-trip (ASYNCcnt).
// Builtin signature (from clang diagnostic): param1 = AS(1) pointer to
// int __attribute__((vector_size(16))), param2 = LDS-side pointer, then two
// immediate ints (offset, cpol). Generic->AS cast via integer round-trip:
// low 32 bits of a generic LDS pointer are the LDS byte offset.
typedef int v4i_b128 __attribute__((vector_size(16)));
typedef __attribute__((address_space(1))) v4i_b128 gv4i_b128;
typedef __attribute__((address_space(3))) v4i_b128 lv4i_b128;

__device__ __forceinline__ void async_copy_b128(const bf16* g, bf16* l) {
  __builtin_amdgcn_global_load_async_to_lds_b128(
      (gv4i_b128*)(unsigned long long)(size_t)g,
      (lv4i_b128*)(unsigned)(size_t)l,
      /*offset=*/0, /*cpol=*/0);
}
__device__ __forceinline__ void wait_async0() {
#if __has_builtin(__builtin_amdgcn_s_wait_asynccnt)
  __builtin_amdgcn_s_wait_asynccnt(0);
#else
  asm volatile("s_wait_asynccnt 0x0" ::: "memory");
#endif
}
#endif

__device__ __forceinline__ v8f wmma_bf16(v16bf a, v16bf b, v8f c) {
  // D = A(16x32 bf16) * B(32x16 bf16) + C(16x16 f32)
  return __builtin_amdgcn_wmma_f32_16x16x32_bf16(
      /*neg_a=*/false, a, /*neg_b=*/false, b,
      /*c_mod=*/(short)0, c, /*reuse_a=*/false, /*reuse_b=*/false);
}

__device__ __forceinline__ v8f zero_v8f() {
  v8f z;
#pragma unroll
  for (int i = 0; i < 8; ++i) z[i] = 0.0f;
  return z;
}

// Load a 16x32 bf16 WMMA operand fragment (A layout; B operand is the mirror
// with "row" = N column index). Row-major source with rowStride elements.
// Lane pattern per ISA: lanes 0-15 hold K 0..7 & 16..23, lanes 16-31 hold
// K 8..15 & 24..31, as two contiguous 16-byte chunks.
__device__ __forceinline__ v16bf load_frag(const bf16* base, int rowStride,
                                           int row, int kbase) {
  const int lane  = threadIdx.x & 31;
  const int khalf = (lane >> 4) * 8;
  const bf16* p = base + (size_t)row * rowStride + kbase + khalf;
  v16bf f;
  uint4* fd = reinterpret_cast<uint4*>(&f);
  fd[0] = *reinterpret_cast<const uint4*>(p);
  fd[1] = *reinterpret_cast<const uint4*>(p + 16);
  return f;
}

// ---------------------------------------------------------------------------
// fp32 -> bf16 conversion
// ---------------------------------------------------------------------------
__global__ void cvt_f32_bf16(const float* __restrict__ in,
                             bf16* __restrict__ out, int n) {
  int i = blockIdx.x * blockDim.x + threadIdx.x;
  if (i < n) out[i] = (bf16)in[i];
}

// ---------------------------------------------------------------------------
// GEMM NT: C[M,N] = A[M,K] * B[N,K]^T, A/B bf16 row-major, f32 accumulate.
// Block = 256 threads (8 waves) computes a 128x128 tile; each wave a 64x32
// subtile (4x2 grid of 16x16 WMMA tiles). K staged through LDS in 64 slices
// (2 WMMA K-steps per barrier pair). M,N multiples of 128, K of 64.
// ---------------------------------------------------------------------------
template <typename OutT>
__global__ __launch_bounds__(256) void gemm_nt_wmma(
    const bf16* __restrict__ A, const bf16* __restrict__ B,
    OutT* __restrict__ C, int M, int N, int K) {
  __shared__ __align__(16) bf16 As[128 * 64];   // 16 KB
  __shared__ __align__(16) bf16 Bs[128 * 64];   // 16 KB

  const int tid   = threadIdx.x;
  const int lane  = tid & 31;
  const int wave  = tid >> 5;   // 0..7
  const int waveM = wave >> 2;  // 0..1 -> 64 rows
  const int waveN = wave & 3;   // 0..3 -> 32 cols
  const int bM    = blockIdx.y * 128;
  const int bN    = blockIdx.x * 128;

  v8f acc[4][2];
#pragma unroll
  for (int i = 0; i < 4; ++i)
#pragma unroll
    for (int j = 0; j < 2; ++j) acc[i][j] = zero_v8f();

  const int srow = tid >> 1;          // 0..127
  const int soff = (tid & 1) * 32;    // element offset; 64B per thread/matrix

  for (int k0 = 0; k0 < K; k0 += 64) {
    // ---- stage 128x64 bf16 tiles of A and B into LDS ----
    const bf16* ga = A + (size_t)(bM + srow) * K + k0 + soff;
    const bf16* gb = B + (size_t)(bN + srow) * K + k0 + soff;
    bf16* la = &As[srow * 64 + soff];
    bf16* lb = &Bs[srow * 64 + soff];
#if USE_ASYNC_LDS
#pragma unroll
    for (int j = 0; j < 4; ++j) {
      async_copy_b128(ga + j * 8, la + j * 8);
      async_copy_b128(gb + j * 8, lb + j * 8);
    }
    if (k0 + 64 < K) {  // prefetch next K slice into L2/L0
      __builtin_prefetch(ga + 64, 0, 1);
      __builtin_prefetch(gb + 64, 0, 1);
    }
    wait_async0();
#else
#pragma unroll
    for (int j = 0; j < 4; ++j) {
      reinterpret_cast<uint4*>(la)[j] = reinterpret_cast<const uint4*>(ga)[j];
      reinterpret_cast<uint4*>(lb)[j] = reinterpret_cast<const uint4*>(gb)[j];
    }
    if (k0 + 64 < K) {
      __builtin_prefetch(ga + 64, 0, 1);
      __builtin_prefetch(gb + 64, 0, 1);
    }
#endif
    __syncthreads();

    // ---- WMMA on two K=32 sub-slices ----
#pragma unroll
    for (int kk = 0; kk < 64; kk += 32) {
      v16bf bfr[2];
#pragma unroll
      for (int tn = 0; tn < 2; ++tn)
        bfr[tn] = load_frag(Bs, 64, waveN * 32 + tn * 16 + (lane & 15), kk);
#pragma unroll
      for (int tm = 0; tm < 4; ++tm) {
        v16bf af = load_frag(As, 64, waveM * 64 + tm * 16 + (lane & 15), kk);
#pragma unroll
        for (int tn = 0; tn < 2; ++tn)
          acc[tm][tn] = wmma_bf16(af, bfr[tn], acc[tm][tn]);
      }
    }
    __syncthreads();
  }

  // ---- write back (C layout: vgpr r, lanes 0-15 -> M=r, lanes 16-31 -> M=r+8)
  const int rbase = 8 * (lane >> 4);
  const int cbase = lane & 15;
#pragma unroll
  for (int tm = 0; tm < 4; ++tm)
#pragma unroll
    for (int tn = 0; tn < 2; ++tn)
#pragma unroll
      for (int r = 0; r < 8; ++r) {
        int row = bM + waveM * 64 + tm * 16 + r + rbase;
        int col = bN + waveN * 32 + tn * 16 + cbase;
        C[(size_t)row * N + col] = (OutT)acc[tm][tn][r];
      }
}

// ---------------------------------------------------------------------------
// RoPE applied in place to the q and k thirds of the bf16 qkv buffer.
// Each thread owns one (d, d+D/2) pair for one (b,t,h) -> race-free.
// ---------------------------------------------------------------------------
__global__ void rope_inplace(bf16* __restrict__ qkv,
                             const float* __restrict__ cosb,
                             const float* __restrict__ sinb,
                             int B, int T, int H, int D, int C) {
  const int half = D / 2;  // 64
  int i = blockIdx.x * blockDim.x + threadIdx.x;
  int total = B * T * H * half;
  if (i >= total) return;
  int dh = i % half;
  int h  = (i / half) % H;
  int t  = (i / (half * H)) % T;
  int b  = i / (half * H * T);

  float c1 = cosb[t * D + dh],        s1 = sinb[t * D + dh];
  float c2 = cosb[t * D + dh + half], s2 = sinb[t * D + dh + half];
  size_t rowbase = ((size_t)(b * T + t)) * (3 * C) + (size_t)h * D;
#pragma unroll
  for (int part = 0; part < 2; ++part) {  // q then k
    bf16* p = qkv + rowbase + (size_t)part * C;
    float lo = (float)p[dh];
    float hi = (float)p[dh + half];
    p[dh]        = (bf16)(lo * c1 - hi * s1);  // x*cos + (-x[d+half])*sin
    p[dh + half] = (bf16)(hi * c2 + lo * s2);  // x*cos + ( x[d-half])*sin
  }
}

// ---------------------------------------------------------------------------
// Flash attention. grid = (T/64, B*H), block = 128 threads (4 waves).
// Each wave owns 16 q rows; the block shares LDS-staged K (row-major, async)
// and V (transposed [D][32]) tiles of 32 k rows. Online softmax, O
// accumulated as WMMA C operand. q/k/v read straight from RoPE'd qkv.
// ---------------------------------------------------------------------------
__global__ __launch_bounds__(128) void flash_attn_wmma(
    const bf16* __restrict__ qkv, bf16* __restrict__ y,
    int B, int T, int H, int D, int C) {
  __shared__ __align__(16) bf16 Kt[32 * 128];      // [k][d]
  __shared__ __align__(16) bf16 Vt[128 * 32];      // [d][k] (transposed)
  __shared__ __align__(16) bf16 Pb[4][16 * 32];    // per-wave P tile

  const int tid  = threadIdx.x;
  const int lane = tid & 31;
  const int wave = tid >> 5;
  const int bh   = blockIdx.y;
  const int b    = bh / H;
  const int h    = bh % H;
  const int qbase = blockIdx.x * 64 + wave * 16;

  const size_t rowStride = (size_t)3 * C;  // qkv elements per t row
  const bf16* qptr = qkv + (size_t)b * T * rowStride + (size_t)h * D;
  const bf16* kptr = qptr + C;
  const bf16* vptr = qptr + 2 * C;

  // ---- load the wave's Q fragments once (4 K=32 slices over D=128) ----
  const int m     = lane & 15;
  const int khalf = (lane >> 4) * 8;
  v16bf qf[4];
#pragma unroll
  for (int s = 0; s < 4; ++s) {
    const bf16* p = qptr + (size_t)(qbase + m) * rowStride + s * 32 + khalf;
    uint4* fd = reinterpret_cast<uint4*>(&qf[s]);
    fd[0] = *reinterpret_cast<const uint4*>(p);
    fd[1] = *reinterpret_cast<const uint4*>(p + 16);
  }

  v8f o[8];
#pragma unroll
  for (int dt = 0; dt < 8; ++dt) o[dt] = zero_v8f();
  float rmax[8], rsum[8];
#pragma unroll
  for (int r = 0; r < 8; ++r) { rmax[r] = -INFINITY; rsum[r] = 0.0f; }

  const float scale = 0.08838834764831845f;  // 1/sqrt(128)

  for (int kb = 0; kb < T; kb += 32) {
    __syncthreads();
    // ---- stage K tile [32][128] (async) and V^T tile [128][32] ----
    {
      const int kr   = tid >> 2;         // 0..31
      const int dseg = (tid & 3) * 32;   // 0,32,64,96
      const bf16* gk = kptr + (size_t)(kb + kr) * rowStride + dseg;
      bf16* lk = &Kt[kr * 128 + dseg];
#if USE_ASYNC_LDS
#pragma unroll
      for (int j = 0; j < 4; ++j) async_copy_b128(gk + j * 8, lk + j * 8);
#else
#pragma unroll
      for (int j = 0; j < 4; ++j)
        reinterpret_cast<uint4*>(lk)[j] = reinterpret_cast<const uint4*>(gk)[j];
#endif
      const bf16* gv = vptr + (size_t)(kb + kr) * rowStride + dseg;
      bf16 tmp[32];
#pragma unroll
      for (int j = 0; j < 4; ++j)
        reinterpret_cast<uint4*>(tmp)[j] = reinterpret_cast<const uint4*>(gv)[j];
#pragma unroll
      for (int j = 0; j < 32; ++j)
        Vt[(dseg + j) * 32 + kr] = tmp[j];
#if USE_ASYNC_LDS
      wait_async0();
#endif
    }
    __syncthreads();

    // ---- S = Q * K^T for a 16x32 block (two 16x16 WMMA tiles) ----
    v8f s0 = zero_v8f(), s1 = zero_v8f();
#pragma unroll
    for (int st = 0; st < 4; ++st) {
      v16bf kf0 = load_frag(Kt, 128, (lane & 15), st * 32);
      v16bf kf1 = load_frag(Kt, 128, 16 + (lane & 15), st * 32);
      s0 = wmma_bf16(qf[st], kf0, s0);
      s1 = wmma_bf16(qf[st], kf1, s1);
    }

    // ---- online softmax (row reductions over the 16-lane C-layout groups) --
    float corr[8];
#pragma unroll
    for (int r = 0; r < 8; ++r) {
      float a0 = s0[r] * scale;
      float a1 = s1[r] * scale;
      float mx = fmaxf(a0, a1);
#pragma unroll
      for (int off = 8; off > 0; off >>= 1)
        mx = fmaxf(mx, __shfl_xor(mx, off, 32));
      float nm = fmaxf(rmax[r], mx);
      corr[r] = __expf(rmax[r] - nm);
      rmax[r] = nm;
      float p0 = __expf(a0 - nm);
      float p1 = __expf(a1 - nm);
      float ps = p0 + p1;
#pragma unroll
      for (int off = 8; off > 0; off >>= 1)
        ps += __shfl_xor(ps, off, 32);
      rsum[r] = rsum[r] * corr[r] + ps;
      s0[r] = p0;
      s1[r] = p1;
    }

    // ---- relayout P (C layout -> A layout) through per-wave LDS ----
    bf16* pb = &Pb[wave][0];
    const int rbase = 8 * (lane >> 4);
#pragma unroll
    for (int r = 0; r < 8; ++r) {
      pb[(r + rbase) * 32 + (lane & 15)]      = (bf16)s0[r];
      pb[(r + rbase) * 32 + 16 + (lane & 15)] = (bf16)s1[r];
    }
    asm volatile("" ::: "memory");  // keep ds stores before the frag loads

    // ---- O = O*corr + P * V ----
#pragma unroll
    for (int dt = 0; dt < 8; ++dt)
#pragma unroll
      for (int r = 0; r < 8; ++r) o[dt][r] *= corr[r];

    v16bf pa = load_frag(pb, 32, (lane & 15), 0);
#pragma unroll
    for (int dt = 0; dt < 8; ++dt) {
      v16bf vf = load_frag(Vt, 32, dt * 16 + (lane & 15), 0);
      o[dt] = wmma_bf16(pa, vf, o[dt]);
    }
  }

  // ---- normalize and write y[b, t, h*D + d] as bf16 ----
  const int rbase = 8 * (lane >> 4);
  const int cbase = lane & 15;
#pragma unroll
  for (int dt = 0; dt < 8; ++dt)
#pragma unroll
    for (int r = 0; r < 8; ++r) {
      int row = qbase + r + rbase;
      int col = h * D + dt * 16 + cbase;
      float val = o[dt][r] / rsum[r];
      y[((size_t)b * T + row) * C + col] = (bf16)val;
    }
}

// ---------------------------------------------------------------------------
// Launch
// ---------------------------------------------------------------------------
extern "C" void kernel_launch(void* const* d_in, const int* in_sizes, int n_in,
                              void* d_out, int out_size, void* d_ws,
                              size_t ws_size, hipStream_t stream) {
  const int B = 4, T = 2048, C = 2048, H = 16, D = 128;

  const float* x      = (const float*)d_in[0];
  const float* cosb   = (const float*)d_in[1];
  const float* sinb   = (const float*)d_in[2];
  const float* W_attn = (const float*)d_in[3];
  const float* W_proj = (const float*)d_in[4];
  float* out = (float*)d_out;

  // workspace layout (bf16 scratch buffers), all 16B aligned
  char* ws = (char*)d_ws;
  size_t off = 0;
  bf16* x_bf  = (bf16*)(ws + off); off += (size_t)B * T * C * 2;       // 32 MB
  bf16* wa_bf = (bf16*)(ws + off); off += (size_t)3 * C * C * 2;       // 24 MB
  bf16* wp_bf = (bf16*)(ws + off); off += (size_t)C * C * 2;           //  8 MB
  bf16* qkv   = (bf16*)(ws + off); off += (size_t)B * T * 3 * C * 2;   // 96 MB
  bf16* y_bf  = (bf16*)(ws + off);                                     // 32 MB

  // 1) fp32 -> bf16 conversions
  {
    int n;
    n = B * T * C;
    cvt_f32_bf16<<<(n + 255) / 256, 256, 0, stream>>>(x, x_bf, n);
    n = 3 * C * C;
    cvt_f32_bf16<<<(n + 255) / 256, 256, 0, stream>>>(W_attn, wa_bf, n);
    n = C * C;
    cvt_f32_bf16<<<(n + 255) / 256, 256, 0, stream>>>(W_proj, wp_bf, n);
  }

  // 2) qkv = x @ W_attn^T   (M=8192, N=6144, K=2048) -> bf16
  gemm_nt_wmma<bf16><<<dim3((3 * C) / 128, (B * T) / 128), 256, 0, stream>>>(
      x_bf, wa_bf, qkv, B * T, 3 * C, C);

  // 3) RoPE in place on q,k thirds of qkv
  {
    int total = B * T * H * (D / 2);
    rope_inplace<<<(total + 255) / 256, 256, 0, stream>>>(
        qkv, cosb, sinb, B, T, H, D, C);
  }

  // 4) flash attention -> y_bf [B,T,C]
  flash_attn_wmma<<<dim3(T / 64, B * H), 128, 0, stream>>>(
      qkv, y_bf, B, T, H, D, C);

  // 5) out = y @ W_proj^T   (M=8192, N=2048, K=2048) -> fp32
  gemm_nt_wmma<float><<<dim3(C / 128, (B * T) / 128), 256, 0, stream>>>(
      y_bf, wp_bf, out, B * T, C, C);
}
